// Model_43654047596636
// MI455X (gfx1250) — compile-verified, matrix operand order checked by reading, not run on hardware
//
#include <hip/hip_runtime.h>
#include <hip/hip_bf16.h>

// Model dims (fixed by reference)
#define HH 4
#define DD 256
#define NN 8192
#define TT 1024
#define VV 256
#define ITERS 6

typedef __bf16 bf16;
typedef __attribute__((ext_vector_type(8)))  __bf16 v8bf;
typedef __attribute__((ext_vector_type(16))) __bf16 v16bf;
typedef __attribute__((ext_vector_type(8)))  float  v8f;
typedef __attribute__((ext_vector_type(4)))  unsigned int u32x4;
typedef __attribute__((ext_vector_type(8)))  unsigned int u32x8;

// ---------------------------------------------------------------------------
// Fragment load (works for both LDS and global pointers):
// CDNA5 16-bit A/B layout for 16x16x32 WMMA.
// lane 0-15 : row = lane,    K = {0..7} in v[0..3], K = {16..23} in v[4..7]
// lane 16-31: row = lane-16, K = {8..15} in v[0..3], K = {24..31} in v[4..7]
// p must already point at (row, k0 + koff) where koff = (lane>>4)*8.
// ---------------------------------------------------------------------------
__device__ __forceinline__ v16bf ldfrag(const bf16* p) {
  v8bf lo = *(const v8bf*)(p);
  v8bf hi = *(const v8bf*)(p + 16);
  v16bf r;
#pragma unroll
  for (int i = 0; i < 8; ++i) { r[i] = lo[i]; r[i + 8] = hi[i]; }
  return r;
}

// ---------------------------------------------------------------------------
// Tensor Data Mover: DMA a 2D bf16 tile (tile_rows x tile_cols elements,
// row stride = stride_elems) from global memory into LDS at lds_off.
// D# per CDNA5 ISA 8.3/8.4: group0 = {count=1, lds_addr, global_addr, type=2},
// group1 = {data_size=2B, tensor_dim = tile_dim (no OOB), dim0 stride}.
// Issued once per wave (scalar op), tracked with TENSORcnt.
// ---------------------------------------------------------------------------
__device__ __forceinline__ void tdm_load_tile_2d(
    unsigned lds_off, const bf16* g, unsigned tile_rows, unsigned tile_cols,
    unsigned stride_elems)
{
  unsigned long long ga = (unsigned long long)(const void*)g;
  u32x4 g0;
  g0[0] = 1u;                                        // count=1, user mode
  g0[1] = lds_off;                                   // lds_addr [63:32]
  g0[2] = (unsigned)ga;                              // global_addr lo
  g0[3] = ((unsigned)(ga >> 32) & 0x01ffffffu) | (2u << 30);  // addr hi, type=2
  u32x8 g1;
  g1[0] = 0x00010000u;                               // data_size = 2 bytes
  g1[1] = (tile_cols & 0xffffu) << 16;               // tensor_dim0 [79:48]
  g1[2] = (tile_cols >> 16) | ((tile_rows & 0xffffu) << 16);  // td0 hi | td1 lo
  g1[3] = (tile_rows >> 16) | (tile_cols << 16);     // td1 hi | tile_dim0
  g1[4] = tile_rows & 0xffffu;                       // tile_dim1 (tile_dim2=0)
  g1[5] = stride_elems;                              // tensor_dim0_stride lo32
  g1[6] = 0u;                                        // stride hi | dim1_stride lo
  g1[7] = 0u;
  u32x4 gz;
  gz[0] = 0u; gz[1] = 0u; gz[2] = 0u; gz[3] = 0u;    // groups 2/3 unused (2D)
  asm volatile("tensor_load_to_lds %0, %1, %2, %3"
               :: "s"(g0), "s"(g1), "s"(gz), "s"(gz)
               : "memory");
}

// Epilogue modes
#define MODE_F32    0
#define MODE_RELU   1
#define MODE_SCORES 2
#define MODE_BIND   3

// ---------------------------------------------------------------------------
// bf16 WMMA GEMM with TDM double-buffered LDS staging.
//   C[M x N] = A[M x K] * Bt[N x K]^T   (Bt row-major = B transposed)
// Block: 256 threads = 8 waves (2x4); block tile 64x128; wave tile 32x32.
// K-step 64: stage A(64x64) + B(128x64) via tensor_load_to_lds, compute 8 WMMA
// per wave per step from ds_load_b128 fragments while next tile DMA overlaps.
// a_hshift/a_hstride: A addressed as base + (k>>a_hshift)*a_hstride +
//                     row*lda + (k - (k>>a_hshift<<a_hshift))  [(T,H*N) reshape]
// ---------------------------------------------------------------------------
__global__ __launch_bounds__(256) void gemm_bf16_wmma(
    const bf16* __restrict__ A, const bf16* __restrict__ Bt,
    float* __restrict__ outF, bf16* __restrict__ outB,
    const bf16* __restrict__ aux,
    int M, int N, int K, int lda, int ldb,
    int a_hshift, long long a_hstride,
    long long zsa, long long zsb, long long zso,
    int mode, float scale)
{
  __shared__ __align__(16) bf16 ldsA[2][64 * 64];
  __shared__ __align__(16) bf16 ldsB[2][128 * 64];

  A  += (long long)blockIdx.z * zsa;
  Bt += (long long)blockIdx.z * zsb;
  if (outF) outF += (long long)blockIdx.z * zso;
  if (outB) outB += (long long)blockIdx.z * zso;
  if (aux)  aux  += (long long)blockIdx.z * zso;

  const int lane = threadIdx.x & 31;
  const int wid  = threadIdx.x >> 5;
  const int wr   = wid >> 2;                 // 0..1
  const int wc   = wid & 3;                  // 0..3
  const int bm   = blockIdx.x * 64;
  const int bn   = blockIdx.y * 128;

  const int r    = lane & 15;
  const int koff = (lane >> 4) << 3;         // 0 or 8

  v8f acc[2][2];
#pragma unroll
  for (int i = 0; i < 2; ++i)
#pragma unroll
    for (int j = 0; j < 2; ++j)
#pragma unroll
      for (int e = 0; e < 8; ++e) acc[i][j][e] = 0.0f;

  // Stage tiles for K block starting at k0 into buffer buf (wave 0 only).
  auto issue = [&](int buf, int k0) {
    const int h  = k0 >> a_hshift;
    const int kk = k0 - (h << a_hshift);
    const bf16* ag = A + (long long)h * a_hstride + (long long)bm * lda + kk;
    const bf16* bg = Bt + (long long)bn * ldb + k0;
    tdm_load_tile_2d((unsigned)(unsigned long long)(const void*)&ldsA[buf][0],
                     ag, 64u, 64u, (unsigned)lda);
    tdm_load_tile_2d((unsigned)(unsigned long long)(const void*)&ldsB[buf][0],
                     bg, 128u, 64u, (unsigned)ldb);
  };

  if (wid == 0) {
    issue(0, 0);
    __builtin_amdgcn_s_wait_tensorcnt(0);
  }
  __syncthreads();

  for (int k0 = 0; k0 < K; k0 += 64) {
    const int cur = (k0 >> 6) & 1;
    const bool more = (k0 + 64) < K;
    if (wid == 0 && more) issue(cur ^ 1, k0 + 64);   // overlap DMA with math

#pragma unroll
    for (int ks = 0; ks < 64; ks += 32) {
      const bf16* ap = &ldsA[cur][(wr * 32 + r) * 64 + ks + koff];
      const bf16* bp = &ldsB[cur][(wc * 32 + r) * 64 + ks + koff];
      v16bf a0 = ldfrag(ap);
      v16bf a1 = ldfrag(ap + 16 * 64);
      v16bf b0 = ldfrag(bp);
      v16bf b1 = ldfrag(bp + 16 * 64);
      acc[0][0] = __builtin_amdgcn_wmma_f32_16x16x32_bf16(false, a0, false, b0, (short)0, acc[0][0], false, false);
      acc[0][1] = __builtin_amdgcn_wmma_f32_16x16x32_bf16(false, a0, false, b1, (short)0, acc[0][1], false, false);
      acc[1][0] = __builtin_amdgcn_wmma_f32_16x16x32_bf16(false, a1, false, b0, (short)0, acc[1][0], false, false);
      acc[1][1] = __builtin_amdgcn_wmma_f32_16x16x32_bf16(false, a1, false, b1, (short)0, acc[1][1], false, false);
    }

    if (wid == 0 && more) __builtin_amdgcn_s_wait_tensorcnt(0);
    __syncthreads();
  }

  // C layout: VGPR e -> row = e + (lane>=16 ? 8 : 0); col = lane & 15
  const int wm    = bm + wr * 32;
  const int wn    = bn + wc * 32;
  const int col0  = lane & 15;
  const int rbase = (lane >> 4) << 3;
#pragma unroll
  for (int i = 0; i < 2; ++i) {
#pragma unroll
    for (int j = 0; j < 2; ++j) {
      const int col = wn + j * 16 + col0;
#pragma unroll
      for (int e = 0; e < 8; ++e) {
        const int row = wm + i * 16 + rbase + e;
        float v = acc[i][j][e];
        const long long idx = (long long)row * N + col;
        if (mode == MODE_F32) {
          outF[idx] = v;
        } else if (mode == MODE_RELU) {
          v = v > 0.0f ? v : 0.0f;
          outB[idx] = (bf16)v;
        } else if (mode == MODE_SCORES) {
          v = (col < row) ? v * scale : 0.0f;   // strict lower triangle
          outB[idx] = (bf16)v;
        } else {                                 // MODE_BIND: relu(y)*xl
          v = v > 0.0f ? v : 0.0f;
          outB[idx] = (bf16)(v * (float)aux[idx]);
        }
      }
    }
  }
}

// ---------------------------------------------------------------------------
// Block-wide sum over 256 threads
// ---------------------------------------------------------------------------
__device__ __forceinline__ float block_sum_256(float v) {
  __shared__ float red[256];
  const int tid = threadIdx.x;
  red[tid] = v;
  __syncthreads();
#pragma unroll
  for (int s = 128; s > 0; s >>= 1) {
    if (tid < s) red[tid] += red[tid + s];
    __syncthreads();
  }
  const float r = red[0];
  __syncthreads();
  return r;
}

// x = LN(emb[ids]); also emit bf16 copies x (T,D) and xT (D,T)
__global__ __launch_bounds__(256) void embed_ln_kernel(
    const int* __restrict__ ids, const float* __restrict__ emb,
    float* __restrict__ x, bf16* __restrict__ xb, bf16* __restrict__ xT)
{
  const int t = blockIdx.x, d = threadIdx.x;
  const float v = emb[(long long)ids[t] * DD + d];
  const float m = block_sum_256(v) * (1.0f / DD);
  const float c = v - m;
  const float var = block_sum_256(c * c) * (1.0f / DD);
  const float o = c * rsqrtf(var + 1e-5f);
  x [t * DD + d] = o;
  xb[t * DD + d] = (bf16)o;
  xT[(long long)d * TT + t] = (bf16)o;
}

// LN per row of (rows, D) f32 -> bf16
__global__ __launch_bounds__(256) void ln_rows_bf16(
    const float* __restrict__ in, bf16* __restrict__ out)
{
  const long long row = blockIdx.x;
  const int d = threadIdx.x;
  const float v = in[row * DD + d];
  const float m = block_sum_256(v) * (1.0f / DD);
  const float c = v - m;
  const float var = block_sum_256(c * c) * (1.0f / DD);
  out[row * DD + d] = (bf16)(c * rsqrtf(var + 1e-5f));
}

// x = LN(x + LN(mlp)); refresh bf16 x and xT
__global__ __launch_bounds__(256) void x_update_kernel(
    const float* __restrict__ mlp, float* __restrict__ x,
    bf16* __restrict__ xb, bf16* __restrict__ xT)
{
  const int t = blockIdx.x, d = threadIdx.x;
  const float v  = mlp[t * DD + d];
  const float m1 = block_sum_256(v) * (1.0f / DD);
  const float c1 = v - m1;
  const float v1 = block_sum_256(c1 * c1) * (1.0f / DD);
  const float l  = c1 * rsqrtf(v1 + 1e-5f);
  const float u  = x[t * DD + d] + l;
  const float m2 = block_sum_256(u) * (1.0f / DD);
  const float c2 = u - m2;
  const float v2 = block_sum_256(c2 * c2) * (1.0f / DD);
  const float o  = c2 * rsqrtf(v2 + 1e-5f);
  x [t * DD + d] = o;
  xb[t * DD + d] = (bf16)o;
  xT[(long long)d * TT + t] = (bf16)o;
}

// qr = xl*cos + rot(xl)*sin ; head pairs (2h,2h+1), freq from n, phase from t
__global__ __launch_bounds__(256) void rotary_kernel(
    const bf16* __restrict__ xl, bf16* __restrict__ qr)
{
  const float TWO_PI = 6.2831853071795864769f;
  const long long total = (long long)HH * TT * NN;
  for (long long i = (long long)blockIdx.x * 256 + threadIdx.x; i < total;
       i += (long long)gridDim.x * 256) {
    const int n = (int)(i & (NN - 1));
    const long long ht = i >> 13;            // NN = 2^13
    const int t = (int)(ht & (TT - 1));
    const int h = (int)(ht >> 10);           // TT = 2^10
    const float xv = (float)xl[i];
    const long long ip = (((long long)((h ^ 1) * TT + t)) << 13) | n;
    const float xo = (float)xl[ip];
    const float rot = (h & 1) ? xo : -xo;
    const float quant = (float)((n >> 1) << 1);
    const float freq = exp2f(-16.0f * quant / (float)NN) * (1.0f / TWO_PI);
    float ph = (float)t * freq;
    ph = (ph - floorf(ph)) * TWO_PI;
    qr[i] = (bf16)(xv * __cosf(ph) + rot * __sinf(ph));
  }
}

// out[z][c][r] = (bf16) in[z][r][c]   (in: Z x R x C)
__global__ __launch_bounds__(256) void transpose_bf16_kernel(
    const float* __restrict__ in, bf16* __restrict__ out, int R, int C, int Z)
{
  const long long total = (long long)Z * R * C;
  for (long long o = (long long)blockIdx.x * 256 + threadIdx.x; o < total;
       o += (long long)gridDim.x * 256) {
    const long long r = o % R;
    const long long tmp = o / R;
    const long long c = tmp % C;
    const long long z = tmp / C;
    out[o] = (bf16)in[(z * R + r) * C + c];
  }
}

// ---------------------------------------------------------------------------
extern "C" void kernel_launch(void* const* d_in, const int* in_sizes, int n_in,
                              void* d_out, int out_size, void* d_ws, size_t ws_size,
                              hipStream_t stream) {
  (void)in_sizes; (void)n_in; (void)out_size; (void)ws_size;
  const int*   ids   = (const int*)  d_in[0];
  const float* emb   = (const float*)d_in[1];
  const float* enc   = (const float*)d_in[2];   // (H, D, N)
  const float* encv  = (const float*)d_in[3];   // (H, D, N)
  const float* dec   = (const float*)d_in[4];   // (H*N, D)
  const float* rdout = (const float*)d_in[5];   // (D, V)
  float* logits = (float*)d_out;                // (T, V) f32

  // Workspace carve-up
  char* w = (char*)d_ws;
  size_t off = 0;
  auto take = [&](size_t bytes) -> void* {
    void* p = w + off;
    off = (off + bytes + 255) & ~(size_t)255;
    return p;
  };
  float* x      = (float*)take((size_t)TT * DD * 4);
  bf16*  xb     = (bf16*) take((size_t)TT * DD * 2);
  bf16*  xT     = (bf16*) take((size_t)TT * DD * 2);
  bf16*  encT   = (bf16*) take((size_t)HH * NN * DD * 2);  // (H,N,D)
  bf16*  encvT  = (bf16*) take((size_t)HH * NN * DD * 2);  // (H,N,D)
  bf16*  decT   = (bf16*) take((size_t)DD * HH * NN * 2);  // (D, H*N)
  bf16*  readT  = (bf16*) take((size_t)VV * DD * 2);       // (V, D)
  bf16*  xl     = (bf16*) take((size_t)HH * TT * NN * 2);  // (H,T,N)
  bf16*  qrbind = (bf16*) take((size_t)HH * TT * NN * 2);  // qr, reused as bind
  bf16*  scores = (bf16*) take((size_t)HH * TT * TT * 2);  // (H,T,T)
  float* attnP  = (float*)take((size_t)HH * TT * DD * 4);  // pre-LN attn
  bf16*  attn   = (bf16*) take((size_t)HH * TT * DD * 2);
  float* mlp    = (float*)take((size_t)TT * DD * 4);

  const dim3 blk(256);
  const long long HS_NONE = 0;
  const float inv_sqrt_n = 0.011048543456039805f;  // 1/sqrt(8192)

  // Weight conversion + transpose to bf16 (N x K row-major operands)
  transpose_bf16_kernel<<<4096, blk, 0, stream>>>(enc,   encT,  DD, NN, HH);
  transpose_bf16_kernel<<<4096, blk, 0, stream>>>(encv,  encvT, DD, NN, HH);
  transpose_bf16_kernel<<<4096, blk, 0, stream>>>(dec,   decT,  HH * NN, DD, 1);
  transpose_bf16_kernel<<<256,  blk, 0, stream>>>(rdout, readT, DD, VV, 1);

  // x = LN(embedding)
  embed_ln_kernel<<<TT, blk, 0, stream>>>(ids, emb, x, xb, xT);

  for (int it = 0; it < ITERS; ++it) {
    // xl[h] = relu(x @ enc[h])        M=1024 N=8192 K=256
    gemm_bf16_wmma<<<dim3(16, 64, HH), blk, 0, stream>>>(
        xb, encT, nullptr, xl, nullptr,
        TT, NN, DD, DD, DD, 31, HS_NONE,
        0, (long long)NN * DD, (long long)TT * NN, MODE_RELU, 0.0f);

    // qr = rotary(xl)
    rotary_kernel<<<4096, blk, 0, stream>>>(xl, qrbind);

    // scores[h] = mask(qr @ qr^T) / sqrt(N)   M=N=1024 K=8192
    gemm_bf16_wmma<<<dim3(16, 8, HH), blk, 0, stream>>>(
        qrbind, qrbind, nullptr, scores, nullptr,
        TT, TT, NN, NN, NN, 31, HS_NONE,
        (long long)TT * NN, (long long)TT * NN, (long long)TT * TT,
        MODE_SCORES, inv_sqrt_n);

    // attnP[h] = scores[h] @ x        M=1024 N=256 K=1024
    gemm_bf16_wmma<<<dim3(16, 2, HH), blk, 0, stream>>>(
        scores, xT, attnP, nullptr, nullptr,
        TT, DD, TT, TT, TT, 31, HS_NONE,
        (long long)TT * TT, 0, (long long)TT * DD, MODE_F32, 0.0f);

    // attn = LN(attnP) per (h,t) row
    ln_rows_bf16<<<HH * TT, blk, 0, stream>>>(attnP, attn);

    // bind[h] = relu(attn[h] @ encv[h]) * xl[h]   M=1024 N=8192 K=256
    gemm_bf16_wmma<<<dim3(16, 64, HH), blk, 0, stream>>>(
        attn, encvT, nullptr, qrbind, xl,
        TT, NN, DD, DD, DD, 31, HS_NONE,
        (long long)TT * DD, (long long)NN * DD, (long long)TT * NN,
        MODE_BIND, 0.0f);

    // mlp = bind(T, H*N) @ decoder    M=1024 N=256 K=32768 (head-blocked A)
    gemm_bf16_wmma<<<dim3(16, 2, 1), blk, 0, stream>>>(
        qrbind, decT, mlp, nullptr, nullptr,
        TT, DD, HH * NN, NN, HH * NN, 13, (long long)TT * NN,
        0, 0, 0, MODE_F32, 0.0f);

    // x = LN(x + LN(mlp))
    x_update_kernel<<<TT, blk, 0, stream>>>(mlp, x, xb, xT);
  }

  // logits = x @ read_out            M=1024 N=256 K=256
  gemm_bf16_wmma<<<dim3(16, 2, 1), blk, 0, stream>>>(
      xb, readT, logits, nullptr, nullptr,
      TT, VV, DD, DD, DD, 31, HS_NONE,
      0, 0, 0, MODE_F32, 0.0f);
}
